// SeparableAttention_17343077941742
// MI455X (gfx1250) — matrix operand order repaired and checked by reading, not compiled, not run on hardware
//
#include <hip/hip_runtime.h>
#include <hip/hip_bf16.h>

// ---------------------------------------------------------------------------
// SeparableAttention fused kernel for gfx1250 (MI455X).
//   B=32, E=256, H=64, W=64, HID=256.
//   One workgroup (256 threads = 8 wave32) per (b,h) row.
//   bf16 WMMA (v_wmma_f32_16x16x32_bf16) with f32 accumulation.
// ---------------------------------------------------------------------------

typedef __attribute__((ext_vector_type(16))) __bf16 v16bf;
typedef __attribute__((ext_vector_type(8)))  float  v8f;

#define E_    256
#define HID_  256
#define H_    64
#define W_    64
#define B_    32
#define HW_   (H_ * W_)

__device__ __forceinline__ unsigned short f32_to_bf16(float f) {
  unsigned int u = __float_as_uint(f);
  u += 0x7FFFu + ((u >> 16) & 1u);   // round-to-nearest-even
  return (unsigned short)(u >> 16);
}

__device__ __forceinline__ v8f wmma_bf16(v16bf a, v16bf b, v8f c) {
  // 8 args: (neg_a, A, neg_b, B, c_mod, C, reuse_a, reuse_b)
  return __builtin_amdgcn_wmma_f32_16x16x32_bf16(false, a, false, b,
                                                 (short)0, c, false, false);
}

// A-fragment (pre-swizzled in workspace): lane reads 32 contiguous bytes.
__device__ __forceinline__ v16bf load_frag_a(const unsigned short* __restrict__ base,
                                             int mt, int kt, int lane) {
  const unsigned short* p = base + (size_t)(mt * 8 + kt) * 512 + lane * 16;
  union { uint4 u[2]; v16bf v; } t;
  t.u[0] = *(const uint4*)(p);
  t.u[1] = *(const uint4*)(p + 8);
  return t.v;
}

// B-fragment from LDS: lane reads 32 contiguous bytes (2x ds_load_b128).
__device__ __forceinline__ v16bf load_frag_b(const unsigned short* frag,
                                             int kt, int nt, int lane) {
  const unsigned short* p = frag + (kt * 4 + nt) * 512 + lane * 16;
  union { uint4 u[2]; v16bf v; } t;
  t.u[0] = *(const uint4*)(p);
  t.u[1] = *(const uint4*)(p + 8);
  return t.v;
}

// Pack bf16 pair for (c even, c+1) at column w into the B-fragment layout.
// ISA 16-bit B 32x16: N = lane%16, K = (v<4 ? 2v : 16+2(v-4)) + 8*(lane/16) + p.
__device__ __forceinline__ void store_pair_bfrag(unsigned int* frag32,
                                                 int c, int w, unsigned int pk) {
  int kt = c >> 5, nt = w >> 4;
  int kk = c & 31, nn = w & 15;
  int lane = (((kk >> 3) & 1) << 4) | nn;
  int v = ((kk & 7) >> 1) + ((kk >> 4) << 2);
  frag32[(((kt * 4 + nt) * 512) + lane * 16 + v * 2) >> 1] = pk;
}

// ---------------------------------------------------------------------------
// Prep: convert Wk / Wv / Wout rows (f32) -> bf16 A-fragment layout in ws.
// ISA 16-bit A 16x32 (MxK): M = lane%16, half = lane/16,
//   K = (v<4 ? 2v : 16+2(v-4)) + 8*half + p.
// Fragment element offset = lane*16 + v*2 + p  (so each lane owns 32B).
// ---------------------------------------------------------------------------
__global__ void prep_weights(const float* __restrict__ Wqkv,
                             const float* __restrict__ Wout,
                             unsigned short* __restrict__ wkf,
                             unsigned short* __restrict__ wvf,
                             unsigned short* __restrict__ wof) {
  int e = blockIdx.x * 256 + threadIdx.x;      // 3 * 65536 total
  int which = e >> 16;
  int idx = e & 65535;                          // 16 mtiles * 8 ktiles * 512
  int mt = idx >> 12;
  int kt = (idx >> 9) & 7;
  int r  = idx & 511;
  int lane = r >> 4;
  int t = r & 15;
  int v = t >> 1;
  int p = t & 1;
  int M  = lane & 15;
  int hf = lane >> 4;
  int K = (v < 4 ? 2 * v : 16 + 2 * (v - 4)) + 8 * hf + p;
  int row = mt * 16 + M;
  int col = kt * 32 + K;

  float s;
  unsigned short* dst;
  if (which == 0)      { s = Wqkv[(size_t)(1 + row) * E_ + col];         dst = wkf; }
  else if (which == 1) { s = Wqkv[(size_t)(1 + HID_ + row) * E_ + col];  dst = wvf; }
  else                 { s = Wout[(size_t)row * HID_ + col];             dst = wof; }
  dst[idx] = f32_to_bf16(s);
}

// ---------------------------------------------------------------------------
// Fused kernel: one workgroup per (b, h) row.
// LDS: [0,32K)  bf16 B-fragments (x, later gated)
//      [32K,96K)  key  f32 256x64
//      [96K,160K) value f32 256x64
//      [160K...)  qbuf[64], sbuf[64], ctx[256]
// ---------------------------------------------------------------------------
__global__ __launch_bounds__(256, 1)
void sepattn_fused(const float* __restrict__ x,
                   const float* __restrict__ Wqkv,
                   const float* __restrict__ bqkv,
                   const float* __restrict__ bout,
                   const unsigned short* __restrict__ wkf,
                   const unsigned short* __restrict__ wvf,
                   const unsigned short* __restrict__ wof,
                   float* __restrict__ out) {
  extern __shared__ char smem[];
  unsigned short* frag   = (unsigned short*)smem;               // 32 KB
  unsigned int*   frag32 = (unsigned int*)smem;
  float* keybuf = (float*)(smem + 32 * 1024);                   // 64 KB
  float* valbuf = (float*)(smem + 96 * 1024);                   // 64 KB
  float* qbuf   = (float*)(smem + 160 * 1024);                  // 64
  float* sbuf   = qbuf + 64;                                    // 64
  float* ctx    = sbuf + 64;                                    // 256

  const int tid  = threadIdx.x;
  const int lane = tid & 31;
  const int wave = tid >> 5;
  const int wg = blockIdx.x;
  const int b = wg >> 6;            // / 64
  const int h = wg & 63;

  const float* xb = x + (size_t)b * E_ * HW_ + (size_t)h * W_;  // + c*HW_ + w
  float*       ob = out + (size_t)b * E_ * HW_ + (size_t)h * W_;

  // ---- Phase 0: stage x tile (256 x 64) as bf16 B-fragments in LDS -------
  {
    const int w0  = tid & 63;
    const int cp0 = tid >> 6;       // 0..3
#pragma unroll 4
    for (int i = 0; i < 32; ++i) {
      int c = (cp0 + i * 4) * 2;    // even channel
      float a0 = xb[(size_t)c * HW_ + w0];
      float a1 = xb[(size_t)(c + 1) * HW_ + w0];
      unsigned int pk = (unsigned int)f32_to_bf16(a0) |
                        ((unsigned int)f32_to_bf16(a1) << 16);
      store_pair_bfrag(frag32, c, w0, pk);
    }
  }
  __syncthreads();

  // ---- Phase 1a: query row in f32 (exact, it's tiny) ---------------------
  if (tid < 64) {
    float acc = bqkv[0];
    for (int c = 0; c < E_; ++c)
      acc += Wqkv[c] * xb[(size_t)c * HW_ + tid];
    qbuf[tid] = acc;
  }

  // ---- Phase 1b: GEMM1  (key|value)(256x64) = W(256x256) @ x(256x64) -----
  {
    const bool iskey = (wave < 4);
    const unsigned short* wbase = iskey ? wkf : wvf;
    float* obuf = iskey ? keybuf : valbuf;
    const float* brow = bqkv + (iskey ? 1 : 1 + HID_);
    const int mbase = (wave & 3) * 4;
    const int hf = lane >> 4, nn = lane & 15;

    for (int np = 0; np < 2; ++np) {
      v8f acc[4][2] = {};
      for (int kt = 0; kt < 8; ++kt) {
        v16bf bf0 = load_frag_b(frag, kt, np * 2 + 0, lane);
        v16bf bf1 = load_frag_b(frag, kt, np * 2 + 1, lane);
#pragma unroll
        for (int m = 0; m < 4; ++m) {
          v16bf af = load_frag_a(wbase, mbase + m, kt, lane);
          acc[m][0] = wmma_bf16(af, bf0, acc[m][0]);
          acc[m][1] = wmma_bf16(af, bf1, acc[m][1]);
        }
      }
      // epilogue: C/D layout -> f32 LDS rows, add bias
#pragma unroll
      for (int m = 0; m < 4; ++m) {
        int cbase = (mbase + m) * 16 + hf * 8;
#pragma unroll
        for (int n = 0; n < 2; ++n) {
          int wcol = (np * 2 + n) * 16 + nn;
#pragma unroll
          for (int r = 0; r < 8; ++r) {
            int c = cbase + r;
            obuf[c * W_ + wcol] = acc[m][n][r] + brow[c];
          }
        }
      }
    }
  }
  __syncthreads();

  // ---- Phase 2: softmax over W of query ----------------------------------
  if (tid < 64) {
    float mx = -1e30f;
    for (int i = 0; i < 64; ++i) mx = fmaxf(mx, qbuf[i]);
    float s = 0.f;
    for (int i = 0; i < 64; ++i) s += __expf(qbuf[i] - mx);
    sbuf[tid] = __expf(qbuf[tid] - mx) / s;
  }
  __syncthreads();

  // ---- Phase 3: context vector ctx[c] = sum_w key[c,w]*score[w] ----------
  {
    int c = tid;
    float s = 0.f;
    for (int w = 0; w < 64; ++w) s += keybuf[c * W_ + w] * sbuf[w];
    ctx[c] = s;
  }
  __syncthreads();

  // ---- Phase 4: gated = relu(value)*ctx -> bf16 B-fragments (reuse) ------
  {
    const int w0  = tid & 63;
    const int cp0 = tid >> 6;
#pragma unroll 4
    for (int i = 0; i < 32; ++i) {
      int c = (cp0 + i * 4) * 2;
      float g0 = fmaxf(valbuf[c * W_ + w0], 0.f) * ctx[c];
      float g1 = fmaxf(valbuf[(c + 1) * W_ + w0], 0.f) * ctx[c + 1];
      unsigned int pk = (unsigned int)f32_to_bf16(g0) |
                        ((unsigned int)f32_to_bf16(g1) << 16);
      store_pair_bfrag(frag32, c, w0, pk);
    }
  }
  __syncthreads();

  // ---- Phase 5: GEMM2  out(256x64) = Wout(256x256) @ gated(256x64) -------
  {
    const int hf = lane >> 4, nn = lane & 15;
    v8f acc[2][4] = {};
    for (int kt = 0; kt < 8; ++kt) {
      v16bf bf[4];
#pragma unroll
      for (int n = 0; n < 4; ++n) bf[n] = load_frag_b(frag, kt, n, lane);
#pragma unroll
      for (int m = 0; m < 2; ++m) {
        v16bf af = load_frag_a(wof, wave * 2 + m, kt, lane);
#pragma unroll
        for (int n = 0; n < 4; ++n) acc[m][n] = wmma_bf16(af, bf[n], acc[m][n]);
      }
    }
#pragma unroll
    for (int m = 0; m < 2; ++m) {
      int obase = (wave * 2 + m) * 16 + hf * 8;
#pragma unroll
      for (int n = 0; n < 4; ++n) {
        int wcol = n * 16 + nn;
#pragma unroll
        for (int r = 0; r < 8; ++r) {
          int o = obase + r;
          ob[(size_t)o * HW_ + wcol] = acc[m][n][r] + bout[o];
        }
      }
    }
  }
}

// ---------------------------------------------------------------------------
extern "C" void kernel_launch(void* const* d_in, const int* in_sizes, int n_in,
                              void* d_out, int out_size, void* d_ws, size_t ws_size,
                              hipStream_t stream) {
  const float* x    = (const float*)d_in[0];
  const float* Wqkv = (const float*)d_in[1];
  const float* bqkv = (const float*)d_in[2];
  const float* Wout = (const float*)d_in[3];
  const float* bout = (const float*)d_in[4];
  float* out = (float*)d_out;

  // workspace: 3 x 64K bf16 fragment images (128 KB each)
  unsigned short* wkf = (unsigned short*)d_ws;
  unsigned short* wvf = wkf + 65536;
  unsigned short* wof = wvf + 65536;

  prep_weights<<<768, 256, 0, stream>>>(Wqkv, Wout, wkf, wvf, wof);

  const size_t lds_bytes = 160 * 1024 + (64 + 64 + 256) * sizeof(float);
  sepattn_fused<<<B_ * H_, 256, lds_bytes, stream>>>(
      x, Wqkv, bqkv, bout, wkf, wvf, wof, out);
}
